// WoodburyRetractionSolver_46265387712742
// MI455X (gfx1250) — matrix operand order repaired and checked by reading, not compiled, not run on hardware
//
#include <hip/hip_runtime.h>
#include <hip/hip_bf16.h>
#include <math.h>

typedef __attribute__((ext_vector_type(16))) __bf16 v16bf;
typedef __attribute__((ext_vector_type(8)))  __bf16 v8bf;
typedef __attribute__((ext_vector_type(8)))  float  v8f;
typedef __attribute__((ext_vector_type(4)))  int    v4i;

#if defined(__has_builtin)
#  if __has_builtin(__builtin_amdgcn_global_load_async_to_lds_b128)
#    define USE_ASYNC_LDS 1
#  endif
#endif
#ifndef USE_ASYNC_LDS
#  define USE_ASYNC_LDS 0
#endif

#if USE_ASYNC_LDS
#  if __has_builtin(__builtin_amdgcn_s_wait_asynccnt)
#    define WAIT_ASYNC_8() __builtin_amdgcn_s_wait_asynccnt(8)
#    define WAIT_ASYNC_0() __builtin_amdgcn_s_wait_asynccnt(0)
#  else
#    define WAIT_ASYNC_8() asm volatile("s_wait_asynccnt 8" ::: "memory")
#    define WAIT_ASYNC_0() asm volatile("s_wait_asynccnt 0" ::: "memory")
#  endif
#else
#  define WAIT_ASYNC_8() ((void)0)
#  define WAIT_ASYNC_0() ((void)0)
#endif

namespace {
constexpr int   kB      = 4096;
constexpr int   kDLat   = 512;
constexpr int   kDModel = 2048;
constexpr int   kRank   = 8;
constexpr int   kNTotal = kDModel * (1 + kRank);   // 18432
constexpr float kDt     = 0.01f;
constexpr float kEps    = 1e-12f;

constexpr int BM = 128, BN = 128, BK = 32;
constexpr int KT = kDLat / BK;                     // 16 k-steps
}

__device__ __forceinline__ float softplus_f(float x) {
    return (x > 0.f) ? (x + log1pf(expf(-x))) : log1pf(expf(x));
}

// one 16B global->LDS transfer (async on CDNA5, sync fallback otherwise)
__device__ __forceinline__ void cp16(const float* __restrict__ g, float* l) {
#if USE_ASYNC_LDS
    __builtin_amdgcn_global_load_async_to_lds_b128(
        (__attribute__((address_space(1))) v4i*)const_cast<float*>(g),
        (__attribute__((address_space(3))) v4i*)l, 0, 0);
#else
    *(float4*)l = *(const float4*)g;
#endif
}

// split 16 f32 into bf16 hi/lo fragment pair
__device__ __forceinline__ void cvt16(const float* fs, v16bf& hi, v16bf& lo) {
    #pragma unroll
    for (int e = 0; e < 16; ++e) {
        const __bf16 h = (__bf16)fs[e];
        hi[e] = h;
        lo[e] = (__bf16)(fs[e] - (float)h);
    }
}

// ---------------------------------------------------------------------------
// Kernel 1: P = z0 @ [W_diag; W_lr]^T + bias via bf16x3 split WMMA.
// Raw f32 tiles double-buffered in LDS via async copies; XOR-swizzled layout
// (float4 chunk ^= row&7) to kill bank aliasing at 128B row stride.
// ---------------------------------------------------------------------------
__global__ __launch_bounds__(256, 1)
void metric_gemm_kernel(const float* __restrict__ z0,
                        const float* __restrict__ Wdiag,
                        const float* __restrict__ bdiag,
                        const float* __restrict__ Wlr,
                        const float* __restrict__ blr,
                        float* __restrict__ Dws,
                        float* __restrict__ Uws)
{
    __shared__ float Af[2][BM * 32];   // 2 x 16KB
    __shared__ float Bf[2][BN * 32];   // 2 x 16KB   (total 64KB)

    const int tid  = threadIdx.x;
    const int lane = tid & 31;
    const int wave = tid >> 5;
    const int wm   = wave & 1;    // 2 wave-rows (64 M each)
    const int wn   = wave >> 1;   // 4 wave-cols (32 N each)
    const int nBase = blockIdx.x * BN;
    const int mBase = blockIdx.y * BM;

    // N blocks never straddle the 2048 boundary (2048 % 128 == 0)
    const float* __restrict__ Wbase =
        (nBase < kDModel) ? (Wdiag + (size_t)nBase * kDLat)
                          : (Wlr   + (size_t)(nBase - kDModel) * kDLat);

    v8f acc[4][2];
    #pragma unroll
    for (int im = 0; im < 4; ++im)
        #pragma unroll
        for (int in = 0; in < 2; ++in)
            #pragma unroll
            for (int e = 0; e < 8; ++e) acc[im][in][e] = 0.f;

    const int ldRow   = tid >> 3;        // 0..31
    const int ldChunk = tid & 7;         // float4 chunk within the 32-float row
    const int rA = lane & 15;
    const int cA = (lane >= 16) ? 2 : 0; // A frag f4-chunks {c, c+1, c+4, c+5} (ISA 7.12.2)
    const int cB = (lane >= 16) ? 2 : 0; // B frag f4-chunks {c, c+1, c+4, c+5}

    auto issueK = [&](int kt) {
        const int k0  = kt * BK;
        const int buf = kt & 1;
        #pragma unroll
        for (int rb = 0; rb < BM; rb += 32) {
            const int row = rb + ldRow;
            const int pc  = (ldChunk ^ (row & 7)) << 2;       // swizzled f32 offset
            cp16(z0    + (size_t)(mBase + row) * kDLat + k0 + (ldChunk << 2),
                 &Af[buf][row * 32 + pc]);
            cp16(Wbase + (size_t)row           * kDLat + k0 + (ldChunk << 2),
                 &Bf[buf][row * 32 + pc]);
        }
    };

    issueK(0);
    for (int kt = 0; kt < KT; ++kt) {
        const bool more = (kt + 1 < KT);
        if (more) issueK(kt + 1);              // 8 more in flight on top of current 8
        if (more) { WAIT_ASYNC_8(); } else { WAIT_ASYNC_0(); }
        __syncthreads();                       // current buffer visible to all waves

        const float* __restrict__ Abuf = Af[kt & 1];
        const float* __restrict__ Bbuf = Bf[kt & 1];

        v16bf afh[4], afl[4], bfh[2], bfl[2];
        #pragma unroll
        for (int im = 0; im < 4; ++im) {
            const int r  = wm * 64 + im * 16 + rA;
            const int rx = r & 7;
            const float* base = Abuf + r * 32;
            const float4 q0 = *(const float4*)(base + (((cA + 0) ^ rx) << 2));
            const float4 q1 = *(const float4*)(base + (((cA + 1) ^ rx) << 2));
            const float4 q2 = *(const float4*)(base + (((cA + 4) ^ rx) << 2));
            const float4 q3 = *(const float4*)(base + (((cA + 5) ^ rx) << 2));
            const float fs[16] = {q0.x,q0.y,q0.z,q0.w, q1.x,q1.y,q1.z,q1.w,
                                  q2.x,q2.y,q2.z,q2.w, q3.x,q3.y,q3.z,q3.w};
            cvt16(fs, afh[im], afl[im]);
        }
        #pragma unroll
        for (int in = 0; in < 2; ++in) {
            const int c  = wn * 32 + in * 16 + rA;
            const int rx = c & 7;
            const float* base = Bbuf + c * 32;
            const float4 q0 = *(const float4*)(base + (((cB + 0) ^ rx) << 2));
            const float4 q1 = *(const float4*)(base + (((cB + 1) ^ rx) << 2));
            const float4 q2 = *(const float4*)(base + (((cB + 4) ^ rx) << 2));
            const float4 q3 = *(const float4*)(base + (((cB + 5) ^ rx) << 2));
            const float fs[16] = {q0.x,q0.y,q0.z,q0.w, q1.x,q1.y,q1.z,q1.w,
                                  q2.x,q2.y,q2.z,q2.w, q3.x,q3.y,q3.z,q3.w};
            cvt16(fs, bfh[in], bfl[in]);
        }

        #pragma unroll
        for (int im = 0; im < 4; ++im)
            #pragma unroll
            for (int in = 0; in < 2; ++in) {
                v8f c = acc[im][in];
                c = __builtin_amdgcn_wmma_f32_16x16x32_bf16(false, afh[im], false, bfh[in], (short)0, c, false, false);
                c = __builtin_amdgcn_wmma_f32_16x16x32_bf16(false, afh[im], false, bfl[in], (short)0, c, false, false);
                c = __builtin_amdgcn_wmma_f32_16x16x32_bf16(false, afl[im], false, bfh[in], (short)0, c, false, false);
                acc[im][in] = c;
            }
        __syncthreads();                       // done reading before next overwrite
    }

    // Epilogue. C/D frag layout: VGPR v, lanes 0-15 -> M=v, lanes 16-31 -> M=v+8, N=lane%16.
    const int rOff = (lane >= 16) ? 8 : 0;
    const int colL = lane & 15;
    #pragma unroll
    for (int in = 0; in < 2; ++in) {
        const int gn = nBase + wn * 32 + in * 16 + colL;
        #pragma unroll
        for (int im = 0; im < 4; ++im)
            #pragma unroll
            for (int v = 0; v < 8; ++v) {
                const int gm = mBase + wm * 64 + im * 16 + rOff + v;
                float val = acc[im][in][v];
                if (gn < kDModel) {
                    Dws[(size_t)gm * kDModel + gn] = softplus_f(val + bdiag[gn]) + 1e-4f;
                } else {
                    const int j = gn - kDModel;
                    Uws[(size_t)gm * (kDModel * kRank) + j] = val + blr[j];
                }
            }
    }
}

// ---------------------------------------------------------------------------
// Kernel 2: per-batch Woodbury solve + midpoint retraction (memory bound).
// ---------------------------------------------------------------------------
__device__ __forceinline__ float waveSum(float v) {
    #pragma unroll
    for (int o = 16; o > 0; o >>= 1) v += __shfl_xor(v, o, 32);
    return v;
}

// core = I + U^T D^-1 U is SPD -> pivot-free elimination, fully unrolled in regs.
__device__ __forceinline__ void solve8(const float* sym /*36 upper*/, const float* rhs, float* out) {
    float M[8][8], y[8], xr[8];
    #pragma unroll
    for (int r = 0; r < 8; ++r) {
        y[r] = rhs[r];
        #pragma unroll
        for (int s = r; s < 8; ++s) {
            float v = sym[r * 8 + s - (r * (r + 1)) / 2];
            if (s == r) v += 1.0f;
            M[r][s] = v; M[s][r] = v;
        }
    }
    #pragma unroll
    for (int k = 0; k < 8; ++k) {
        const float inv = 1.0f / M[k][k];
        #pragma unroll
        for (int i = k + 1; i < 8; ++i) {
            const float f = M[i][k] * inv;
            #pragma unroll
            for (int j = k + 1; j < 8; ++j) M[i][j] -= f * M[k][j];
            y[i] -= f * y[k];
        }
    }
    #pragma unroll
    for (int k = 7; k >= 0; --k) {
        float t = y[k];
        #pragma unroll
        for (int j = k + 1; j < 8; ++j) t -= M[k][j] * xr[j];
        xr[k] = t / M[k][k];
    }
    #pragma unroll
    for (int r = 0; r < 8; ++r) out[r] = xr[r];
}

__global__ __launch_bounds__(256, 1)
void woodbury_retract_kernel(const float* __restrict__ Dws,
                             const float* __restrict__ Uws,
                             const float* __restrict__ h,
                             float* __restrict__ out)
{
    // 0..35 core(sym upper), 36..43 rhs1, 44..51 rhs2, 52 |h|^2, 53 |x|^2, 54 |x2|^2
    __shared__ float s_acc[56];
    __shared__ float s_corr[8], s_corr2[8];
    __shared__ float s_c1, s_c2;

    const int tid  = threadIdx.x;
    const int lane = tid & 31;
    const int b    = blockIdx.x;

    if (tid < 56) s_acc[tid] = 0.f;
    __syncthreads();

    float dinv[8], hv[8], u[8][8];
    float cacc[36], r1[8], hn2 = 0.f;
    #pragma unroll
    for (int t = 0; t < 36; ++t) cacc[t] = 0.f;
    #pragma unroll
    for (int r = 0; r < 8; ++r) r1[r] = 0.f;

    #pragma unroll
    for (int i = 0; i < 8; ++i) {
        const int d = tid + i * 256;
        const size_t base = (size_t)b * kDModel + d;
        dinv[i] = 1.0f / Dws[base];
        const float hvv = h[base];
        hv[i] = hvv;
        hn2 += hvv * hvv;
        const float4* up = (const float4*)(Uws + base * kRank);
        const float4 u0 = up[0], u1 = up[1];
        u[i][0]=u0.x; u[i][1]=u0.y; u[i][2]=u0.z; u[i][3]=u0.w;
        u[i][4]=u1.x; u[i][5]=u1.y; u[i][6]=u1.z; u[i][7]=u1.w;
        const float w = dinv[i];
        #pragma unroll
        for (int r = 0; r < 8; ++r) {
            const float uw = u[i][r] * w;
            r1[r] += uw * hvv;
            #pragma unroll
            for (int s = r; s < 8; ++s)
                cacc[r * 8 + s - (r * (r + 1)) / 2] += uw * u[i][s];
        }
    }

    #pragma unroll
    for (int t = 0; t < 36; ++t) { const float v = waveSum(cacc[t]); if (lane == 0) atomicAdd(&s_acc[t], v); }
    #pragma unroll
    for (int r = 0; r < 8; ++r)  { const float v = waveSum(r1[r]);  if (lane == 0) atomicAdd(&s_acc[36 + r], v); }
    { const float v = waveSum(hn2); if (lane == 0) atomicAdd(&s_acc[52], v); }
    __syncthreads();

    if (tid == 0) solve8(s_acc, s_acc + 36, s_corr);
    __syncthreads();

    float x[8]; float nx = 0.f;
    #pragma unroll
    for (int i = 0; i < 8; ++i) {
        float dotc = 0.f;
        #pragma unroll
        for (int r = 0; r < 8; ++r) dotc += u[i][r] * s_corr[r];
        x[i] = dinv[i] * (hv[i] - dotc);
        nx += x[i] * x[i];
    }
    nx = waveSum(nx);
    if (lane == 0) atomicAdd(&s_acc[53], nx);
    __syncthreads();

    if (tid == 0) s_c1 = 0.5f * kDt * sqrtf(s_acc[52]) / fmaxf(sqrtf(s_acc[53]), kEps);
    __syncthreads();

    const float c1 = s_c1;
    float r2a[8];
    #pragma unroll
    for (int r = 0; r < 8; ++r) r2a[r] = 0.f;
    #pragma unroll
    for (int i = 0; i < 8; ++i) {
        const float w = dinv[i] * (hv[i] + c1 * x[i]);
        #pragma unroll
        for (int r = 0; r < 8; ++r) r2a[r] += u[i][r] * w;
    }
    #pragma unroll
    for (int r = 0; r < 8; ++r) { const float v = waveSum(r2a[r]); if (lane == 0) atomicAdd(&s_acc[44 + r], v); }
    __syncthreads();

    if (tid == 0) solve8(s_acc, s_acc + 44, s_corr2);
    __syncthreads();

    float x2[8]; float nx2 = 0.f;
    #pragma unroll
    for (int i = 0; i < 8; ++i) {
        const float hm = hv[i] + c1 * x[i];
        float dotc = 0.f;
        #pragma unroll
        for (int r = 0; r < 8; ++r) dotc += u[i][r] * s_corr2[r];
        x2[i] = dinv[i] * (hm - dotc);
        nx2 += x2[i] * x2[i];
    }
    nx2 = waveSum(nx2);
    if (lane == 0) atomicAdd(&s_acc[54], nx2);
    __syncthreads();

    if (tid == 0) s_c2 = kDt * sqrtf(s_acc[52]) / fmaxf(sqrtf(s_acc[54]), kEps);
    __syncthreads();

    const float c2 = s_c2;
    #pragma unroll
    for (int i = 0; i < 8; ++i) {
        const int d = tid + i * 256;
        out[(size_t)b * kDModel + d] = hv[i] + c2 * x2[i];
    }
}

extern "C" void kernel_launch(void* const* d_in, const int* in_sizes, int n_in,
                              void* d_out, int out_size, void* d_ws, size_t ws_size,
                              hipStream_t stream) {
    (void)in_sizes; (void)n_in; (void)out_size; (void)ws_size;
    const float* z0    = (const float*)d_in[0];
    const float* h     = (const float*)d_in[1];
    const float* Wdiag = (const float*)d_in[2];
    const float* bdiag = (const float*)d_in[3];
    const float* Wlr   = (const float*)d_in[4];
    const float* blr   = (const float*)d_in[5];
    float* out = (float*)d_out;

    float* Dws = (float*)d_ws;                      // 4096*2048 f32  (32 MiB)
    float* Uws = Dws + (size_t)kB * kDModel;        // 4096*16384 f32 (256 MiB)

    dim3 gridA(kNTotal / BN, kB / BM);              // 144 x 32
    metric_gemm_kernel<<<gridA, 256, 0, stream>>>(z0, Wdiag, bdiag, Wlr, blr, Dws, Uws);
    woodbury_retract_kernel<<<dim3(kB), 256, 0, stream>>>(Dws, Uws, h, out);
}
